// BaseAttentionPooling_88665304859109
// MI455X (gfx1250) — matrix operand order repaired and checked by reading, not compiled, not run on hardware
//
#include <hip/hip_runtime.h>

// ---------------------------------------------------------------------------
// MI455X (gfx1250) fused attention pooling.
// Roofline: x = 512 MB fp32 dominates; 23.3 TB/s HBM => ~22us for ONE pass.
// Strategy: single streaming pass over x. Since |logit| <= ~11.5, exp() is
// safe without max-subtraction, so softmax numerator/denominator fold into
// one segmented accumulation fused right behind the WMMA logit computation.
// GEMM uses v_wmma_f32_16x16x32_f16 (f16 inputs, f32 accumulate).
// This rev: hardware V_TANH_F32 via __builtin_amdgcn_tanhf (guarded).
// ---------------------------------------------------------------------------

typedef __attribute__((ext_vector_type(16))) _Float16 v16h;
typedef __attribute__((ext_vector_type(8)))  float    v8f;
typedef __attribute__((ext_vector_type(4)))  int      i4x;
typedef __attribute__((ext_vector_type(2)))  int      i2x;
typedef __attribute__((ext_vector_type(4)))  float    f4x;

#define D_DIM   256          // input dim
#define H_DIM   128          // hidden dim
#define ROWS    112          // rows (nodes) per block chunk (7 WMMA row-tiles)
#define STRIDE  264          // LDS row stride in halves (16B aligned, bank-spread)
#define TILES   (ROWS / 16)  // 7 row-tiles -> waves 0..6 compute, wave 7 helps ldst

union Frag16 { i4x q[2]; v16h v; };   // 32 bytes = 16 halves = one WMMA operand/lane

__device__ __forceinline__ float fast_tanh(float v) {
#if __has_builtin(__builtin_amdgcn_tanhf)
  return __builtin_amdgcn_tanhf(v);    // single V_TANH_F32 on gfx1250
#else
  return tanhf(v);
#endif
}

// W1 [D][H] fp32 (row-major) -> W1T [H][D] f16, so B-fragment lane loads are
// two contiguous 8-half runs (two 16B reads) per the 32x16 B layout.
__global__ __launch_bounds__(256) void convert_w1_kernel(
    const float* __restrict__ W1, unsigned short* __restrict__ W1T) {
  int idx = blockIdx.x * 256 + threadIdx.x;          // over H*D = 32768
  if (idx < D_DIM * H_DIM) {
    int n = idx / D_DIM;                             // W1T row (hidden unit)
    int k = idx % D_DIM;                             // W1T col (input dim)
    union { _Float16 h; unsigned short u; } cv;
    cv.h = (_Float16)W1[(long)k * H_DIM + n];
    W1T[idx] = cv.u;
  }
}

__global__ __launch_bounds__(256) void fused_attn_pool_kernel(
    const float*     __restrict__ x,      // [N, 256] fp32
    const long long* __restrict__ batch,  // [N] sorted graph ids
    const unsigned short* __restrict__ W1T, // [128, 256] f16 bits
    const float*     __restrict__ b1,     // [128]
    const float*     __restrict__ W2,     // [128]
    const float*     __restrict__ b2,     // [1]
    float*           __restrict__ out,    // [G, 256] unnormalized accumulator
    float*           __restrict__ denom,  // [G]
    int Nn) {
  __shared__ _Float16 xt[ROWS * STRIDE];  // 59,136 B f16 copy of the chunk
  __shared__ float    logits_s[ROWS];
  __shared__ float    e_s[ROWS];
  __shared__ int      batch_s[ROWS];

  const int tid  = threadIdx.x;
  const int lane = tid & 31;
  const int wave = tid >> 5;
  const int r0   = blockIdx.x * ROWS;
  int nr = Nn - r0; if (nr > ROWS) nr = ROWS; if (nr < 0) nr = 0;

  // ---- stage 1: stream chunk of x (fp32, coalesced 16B) -> LDS as f16 ----
  for (int idx4 = tid * 4; idx4 < ROWS * D_DIM; idx4 += 256 * 4) {
    const int row = idx4 >> 8;
    const int col = idx4 & (D_DIM - 1);
    f4x v = {0.f, 0.f, 0.f, 0.f};
    const long grow = (long)r0 + row;
    if (grow < Nn) v = *(const f4x*)(x + grow * D_DIM + col);
    union { _Float16 h[4]; i2x p; } pk;
    pk.h[0] = (_Float16)v.x; pk.h[1] = (_Float16)v.y;
    pk.h[2] = (_Float16)v.z; pk.h[3] = (_Float16)v.w;
    *(i2x*)(&xt[row * STRIDE + col]) = pk.p;
  }
  if (tid < ROWS) {
    const long grow = (long)r0 + tid;
    batch_s[tid] = (grow < Nn) ? (int)batch[grow] : -1;
  }
  __syncthreads();

  // ---- stage 2: logits via WMMA. wave w owns rows [16w, 16w+16) ----
  if (wave < TILES) {
    const int mrow = 16 * wave + (lane & 15);
    const int krun = (lane >> 4) << 3;   // A layout: lanes 16-31 start at K+8
    Frag16 a[8];
#pragma unroll
    for (int k = 0; k < 8; ++k) {        // preload all K fragments of A (K=256)
      const _Float16* ap = &xt[mrow * STRIDE + 32 * k + krun];
      a[k].q[0] = *(const i4x*)ap;          // K runs [k0, k0+8)
      a[k].q[1] = *(const i4x*)(ap + 16);   // and [k0+16, k0+24)
    }
    float acc[8];
#pragma unroll
    for (int i = 0; i < 8; ++i) acc[i] = 0.f;

#pragma unroll
    for (int n = 0; n < 8; ++n) {        // 8 N-tiles cover H=128
      const int   ncol = n * 16 + (lane & 15);
      const float bias = b1[ncol];
      const float w2v  = W2[ncol];
      v8f c;
#pragma unroll
      for (int i = 0; i < 8; ++i) c[i] = bias;   // fold b1 into accumulator
      // B layout (32x16): lane holds N = lane%16, K runs [8h,8h+8) and [16+8h,24+8h)
      const unsigned short* bbase =
          W1T + (long)ncol * D_DIM + ((lane >> 4) << 3);
#pragma unroll
      for (int k = 0; k < 8; ++k) {
        Frag16 b;
        b.q[0] = *(const i4x*)(bbase + 32 * k);
        b.q[1] = *(const i4x*)(bbase + 32 * k + 16);
        c = __builtin_amdgcn_wmma_f32_16x16x32_f16(
            false, a[k].v, false, b.v, (short)0, c, false, false);
      }
      // logit partial: sum over these 16 hidden cols of tanh(h)*W2
#pragma unroll
      for (int i = 0; i < 8; ++i) {
        float t = fast_tanh(c[i]) * w2v;
        t += __shfl_xor(t, 1);
        t += __shfl_xor(t, 2);
        t += __shfl_xor(t, 4);
        t += __shfl_xor(t, 8);           // reduce within each 16-lane half
        acc[i] += t;
      }
    }
    if ((lane & 15) == 0) {              // lane 0 -> rows +0..7, lane 16 -> +8..15
      const float b2v  = b2[0];
      const int   rbase = 16 * wave + ((lane >> 4) << 3);
#pragma unroll
      for (int i = 0; i < 8; ++i) logits_s[rbase + i] = acc[i] + b2v;
    }
  }
  __syncthreads();

  // ---- stage 3: e = exp(logit). Bounded (|logit|<=~11.5) so no max needed.
  if (tid < ROWS) e_s[tid] = (tid < nr) ? __expf(logits_s[tid]) : 0.f;
  __syncthreads();

  // ---- stage 4: segmented accumulation. thread t owns column t.
  // Re-read x fp32 from global (chunk just streamed -> guaranteed L2 hit).
  if (nr > 0) {
    const int col = tid;
    float acc = 0.f;
    int gcur = batch_s[0];
    for (int i = 0; i < nr; ++i) {
      const int g = batch_s[i];
      if (g != gcur) {
        atomicAdd(&out[(long)gcur * D_DIM + col], acc);
        acc = 0.f; gcur = g;
      }
      acc += x[((long)r0 + i) * D_DIM + col] * e_s[i];
    }
    atomicAdd(&out[(long)gcur * D_DIM + col], acc);

    if (tid == 0) {                      // denominator: same walk over e
      float ea = 0.f; int gc = batch_s[0];
      for (int i = 0; i < nr; ++i) {
        const int g = batch_s[i];
        if (g != gc) { atomicAdd(&denom[gc], ea); ea = 0.f; gc = g; }
        ea += e_s[i];
      }
      atomicAdd(&denom[gc], ea);
    }
  }
}

// pooled = accum / denom; empty graphs (denom==0) -> zeros (matches reference).
__global__ __launch_bounds__(256) void normalize_kernel(
    float* __restrict__ out, const float* __restrict__ denom) {
  const int g = blockIdx.x, c = threadIdx.x;
  const float d = denom[g];
  const float v = out[(long)g * D_DIM + c];
  out[(long)g * D_DIM + c] = (d > 0.f) ? (v / d) : 0.f;
}

extern "C" void kernel_launch(void* const* d_in, const int* in_sizes, int n_in,
                              void* d_out, int out_size, void* d_ws, size_t ws_size,
                              hipStream_t stream) {
  const float*     x     = (const float*)d_in[0];
  const long long* batch = (const long long*)d_in[1];
  const float*     W1    = (const float*)d_in[2];
  const float*     b1    = (const float*)d_in[3];
  const float*     W2    = (const float*)d_in[4];
  const float*     b2    = (const float*)d_in[5];
  float* out = (float*)d_out;

  const int Nn = in_sizes[1];            // number of nodes
  const int G  = out_size / D_DIM;       // number of graphs

  float* denom = (float*)d_ws;                                   // [G] f32
  unsigned short* W1T =
      (unsigned short*)((char*)d_ws + (((size_t)G * 4 + 255) & ~(size_t)255));

  // accumulators must start at zero every call (harness poisons buffers)
  hipMemsetAsync(d_out, 0, (size_t)out_size * sizeof(float), stream);
  hipMemsetAsync(denom, 0, (size_t)G * sizeof(float), stream);

  convert_w1_kernel<<<(D_DIM * H_DIM + 255) / 256, 256, 0, stream>>>(W1, W1T);

  const int nblocks = (Nn + ROWS - 1) / ROWS;
  fused_attn_pool_kernel<<<nblocks, 256, 0, stream>>>(
      x, batch, W1T, b1, W2, b2, out, denom, Nn);

  normalize_kernel<<<G, 256, 0, stream>>>(out, denom);
}
